// FlashRWLargeAttention_47399259079206
// MI455X (gfx1250) — matrix operand order repaired and checked by reading, not compiled, not run on hardware
//
#include <hip/hip_runtime.h>
#include <hip/hip_bf16.h>
#include <cstddef>
#include <cstdint>

// ---------------- CDNA5 WMMA types ----------------
typedef __bf16 bf16_t;
typedef __attribute__((ext_vector_type(16))) __bf16 v16bf;
typedef __attribute__((ext_vector_type(8)))  float  v8f;
typedef __attribute__((ext_vector_type(4)))  int    v4i;

union AF { v16bf v; uint4 q[2]; };           // 32 bytes = one bf16 A/B fragment
static_assert(sizeof(v16bf) == 32, "v16bf size");

__device__ __forceinline__ v8f wmma_bf16(v16bf a, v16bf b, v8f c) {
  // D = A(16x32 bf16) * B(32x16 bf16) + C(16x16 f32)
  return __builtin_amdgcn_wmma_f32_16x16x32_bf16(false, a, false, b, (short)0, c,
                                                 false, false);
}

// ---------------- gfx1250 async global->LDS copy (ASYNCcnt path) ----------------
typedef __attribute__((address_space(1))) v4i gv4i;   // global
typedef __attribute__((address_space(3))) v4i lv4i;   // LDS

__device__ __forceinline__ void async_copy_b128(void* lds, const void* g) {
#if __has_builtin(__builtin_amdgcn_global_load_async_to_lds_b128)
  __builtin_amdgcn_global_load_async_to_lds_b128((gv4i*)g, (lv4i*)lds, 0, 0);
#else
  asm volatile("global_load_async_to_lds_b128 %0, %1, off"
               :
               : "v"((uint32_t)(uintptr_t)(__attribute__((address_space(3))) char*)lds),
                 "v"(g)
               : "memory");
#endif
}

__device__ __forceinline__ void wait_async0() {
#if __has_builtin(__builtin_amdgcn_s_wait_asynccnt)
  __builtin_amdgcn_s_wait_asynccnt(0);
#else
  asm volatile("s_wait_asynccnt 0x0" ::: "memory");
#endif
}

// ---------------- constants ----------------
#define T_TOK 4096
#define HID   2048
#define QKVN  3072
#define ATTD  2048
#define SEQL  1024
#define NG    8
#define NH    4

// ---------------- f32 -> bf16 convert (activations) ----------------
__global__ __launch_bounds__(256) void cvt_f32_bf16(const float* __restrict__ src,
                                                    bf16_t* __restrict__ dst, int n4) {
  int i = blockIdx.x * 256 + threadIdx.x;
  if (i < n4) {
    float4 f = ((const float4*)src)[i];
    union { bf16_t h[4]; uint2 u; } o;
    o.h[0] = (bf16_t)f.x; o.h[1] = (bf16_t)f.y;
    o.h[2] = (bf16_t)f.z; o.h[3] = (bf16_t)f.w;
    ((uint2*)dst)[i] = o.u;
  }
}

// ---------------- f32 [K][N] -> bf16 [N][K] transpose (weights, one-time) ----------
__global__ __launch_bounds__(256) void cvt_transpose(const float* __restrict__ src,
                                                     bf16_t* __restrict__ dst,
                                                     int K, int N) {
  __shared__ float tile[32][33];
  const int k0 = blockIdx.y * 32, n0 = blockIdx.x * 32;
  const int tx = threadIdx.x & 31, ty = threadIdx.x >> 5;  // 32 x 8
#pragma unroll
  for (int r = 0; r < 4; ++r) {
    int k = ty + r * 8;
    tile[k][tx] = src[(size_t)(k0 + k) * N + n0 + tx];
  }
  __syncthreads();
#pragma unroll
  for (int r = 0; r < 4; ++r) {
    int n = ty + r * 8;
    dst[(size_t)(n0 + n) * K + k0 + tx] = (bf16_t)tile[tx][n];
  }
}

// ---------------- bf16 WMMA GEMM: C = A(MxK) * Bt(NxK)^T + bias ----------------
// block tile 128x128, K-step 32, 256 threads = 8 waves, async double-buffered LDS.
template <int OUT_BF16>
__global__ __launch_bounds__(256) void gemm_bias(const bf16_t* __restrict__ A,
                                                 const bf16_t* __restrict__ Bt,
                                                 const float* __restrict__ bias,
                                                 void* __restrict__ outp,
                                                 int M, int N, int K) {
  constexpr int LDT = 40;                       // 80B row stride: aligned, conflict-free
  __shared__ bf16_t sA[2][128 * LDT];
  __shared__ bf16_t sB[2][128 * LDT];
  const int tid = threadIdx.x;
  const int wave = tid >> 5, lane = tid & 31;
  const int lh = lane & 15, hv = lane >> 4;
  const int m0 = blockIdx.y * 128, n0 = blockIdx.x * 128;

  auto stage = [&](int buf, int k0) {
#pragma unroll
    for (int r = 0; r < 2; ++r) {
      int idx = tid + r * 256;                  // 0..511
      int row = idx >> 2, ch = idx & 3;
      async_copy_b128(&sA[buf][row * LDT + ch * 8],
                      A + (size_t)(m0 + row) * K + k0 + ch * 8);
      async_copy_b128(&sB[buf][row * LDT + ch * 8],
                      Bt + (size_t)(n0 + row) * K + k0 + ch * 8);
    }
  };

  v8f acc[8] = {};
  stage(0, 0);
  int cur = 0;
  for (int k0 = 0; k0 < K; k0 += 32) {
    wait_async0();
    __syncthreads();
    if (k0 + 32 < K) stage(cur ^ 1, k0 + 32);

    // A fragment: row = lane&15; lanes<16 hold K 0..7 & 16..23, lanes>=16 hold 8..15 & 24..31
    AF a;
    {
      int arow = wave * 16 + lh;
      int koff = hv ? 8 : 0;
      a.q[0] = *(const uint4*)&sA[cur][arow * LDT + koff];
      a.q[1] = *(const uint4*)&sA[cur][arow * LDT + koff + 16];
    }
    // B fragments: column = lane&15; lanes<16 hold K 0..7 & 16..23, lanes>=16 hold 8..15 & 24..31
#pragma unroll
    for (int j = 0; j < 8; ++j) {
      AF b;
      int col = j * 16 + lh;
      int klo = hv ? 8 : 0;
      b.q[0] = *(const uint4*)&sB[cur][col * LDT + klo];
      b.q[1] = *(const uint4*)&sB[cur][col * LDT + klo + 16];
      acc[j] = wmma_bf16(a.v, b.v, acc[j]);
    }
    cur ^= 1;
  }

  // epilogue: C element i -> row = i + hv*8, col = lane&15
#pragma unroll
  for (int j = 0; j < 8; ++j) {
    int col = n0 + j * 16 + lh;
    float bv = bias[col];
#pragma unroll
    for (int i = 0; i < 8; ++i) {
      int row = m0 + wave * 16 + i + hv * 8;
      float v = acc[j][i] + bv;
      if (OUT_BF16)
        ((bf16_t*)outp)[(size_t)row * N + col] = (bf16_t)v;
      else
        ((float*)outp)[(size_t)row * N + col] = v;
    }
  }
}

// ---------------- RoPE + scatter into attention-friendly layouts ----------------
// qkv: [T][G][6][64] bf16 -> Q[b][g][h][s][64], K[b][g][s][64], Vt[b][g][64][s]
__global__ __launch_bounds__(256) void rope_scatter(const bf16_t* __restrict__ qkv,
                                                    const float* __restrict__ cosb,
                                                    const float* __restrict__ sinb,
                                                    bf16_t* __restrict__ Q,
                                                    bf16_t* __restrict__ Kb,
                                                    bf16_t* __restrict__ Vt) {
  const int t = blockIdx.x;
  const int b = t >> 10, s = t & 1023;
  for (int p = threadIdx.x; p < NG * 6 * 32; p += 256) {
    int g = p / 192;
    int r = (p / 32) % 6;
    int j = p & 31;
    const bf16_t* src = qkv + (size_t)t * QKVN + (g * 6 + r) * 64;
    float x1 = (float)src[j], x2 = (float)src[j + 32];
    if (r < 5) {  // rope on q heads and k (row 4); v (row 5) passes through
      float c = cosb[t * 32 + j], sn = sinb[t * 32 + j];
      float o1 = x1 * c - x2 * sn;
      float o2 = x1 * sn + x2 * c;
      x1 = o1; x2 = o2;
    }
    if (r < 4) {
      bf16_t* dst = Q + ((((size_t)b * NG + g) * NH + r) * SEQL + s) * 64;
      dst[j] = (bf16_t)x1; dst[j + 32] = (bf16_t)x2;
    } else if (r == 4) {
      bf16_t* dst = Kb + (((size_t)b * NG + g) * SEQL + s) * 64;
      dst[j] = (bf16_t)x1; dst[j + 32] = (bf16_t)x2;
    } else {
      bf16_t* dst = Vt + (((size_t)b * NG + g) * 64) * SEQL;
      dst[(size_t)j * SEQL + s]        = (bf16_t)x1;
      dst[(size_t)(j + 32) * SEQL + s] = (bf16_t)x2;
    }
  }
}

// ---------------- flash attention (causal, GQA) ----------------
// block = (b, g, 32-row m-pair); 8 waves: wave w -> head (w&3), m-tile mp*32 + (w>>2)*16.
// K/V tiles async double-buffered in LDS, shared by all 8 waves.
__global__ __launch_bounds__(256) void attn_kernel(const bf16_t* __restrict__ Q,
                                                   const bf16_t* __restrict__ Kb,
                                                   const bf16_t* __restrict__ Vt,
                                                   bf16_t* __restrict__ Out) {
  __shared__ bf16_t sK[2][32 * 64];       // [t][d]
  __shared__ bf16_t sV[2][64 * 32];       // [d][t]
  __shared__ bf16_t sP[8 * 16 * 32];      // per-wave P staging (C-layout -> A-layout)
  const int tid = threadIdx.x, wave = tid >> 5, lane = tid & 31;
  const int lh = lane & 15, hv = lane >> 4;
  const int mp = blockIdx.x & 31;
  const int bg = blockIdx.x >> 5;
  const int g = bg & 7, b = bg >> 3;
  const int head = wave & 3;
  const int mrow = mp * 32 + (wave >> 2) * 16;
  const float scale = 0.125f;              // 64^-0.5

  // Q fragments (A-layout) straight from global: row = mrow + (lane&15)
  const bf16_t* Qbase =
      Q + ((((size_t)b * NG + g) * NH + head) * SEQL + (mrow + lh)) * 64;
  AF aq[2];
  {
    int koff = hv ? 8 : 0;
    aq[0].q[0] = *(const uint4*)(Qbase + koff);
    aq[0].q[1] = *(const uint4*)(Qbase + koff + 16);
    aq[1].q[0] = *(const uint4*)(Qbase + 32 + koff);
    aq[1].q[1] = *(const uint4*)(Qbase + 32 + koff + 16);
  }
  const bf16_t* Kbg = Kb + (((size_t)b * NG + g) * SEQL) * 64;
  const bf16_t* Vbg = Vt + (((size_t)b * NG + g) * 64) * SEQL;

  auto stageKV = [&](int buf, int t0) {
    {
      int row = tid >> 3, ch = tid & 7;
      async_copy_b128(&sK[buf][row * 64 + ch * 8],
                      Kbg + (size_t)(t0 + row) * 64 + ch * 8);
    }
    {
      int d = tid >> 2, ch = tid & 3;
      async_copy_b128(&sV[buf][d * 32 + ch * 8],
                      Vbg + (size_t)d * SEQL + t0 + ch * 8);
    }
  };

  float mi[8], li[8];
  v8f acc[4] = {};
#pragma unroll
  for (int i = 0; i < 8; ++i) { mi[i] = -3.0e38f; li[i] = 0.f; }

  const int tend = mp * 32 + 31;
  stageKV(0, 0);
  int cur = 0;
  for (int t0 = 0; t0 <= tend; t0 += 32) {
    wait_async0();
    __syncthreads();
    if (t0 + 32 <= tend) stageKV(cur ^ 1, t0 + 32);

    if (t0 <= mrow + 15) {
      const bf16_t* kb = sK[cur];
      const bf16_t* vb = sV[cur];
      // scores: Q(16x64) * K^T(64x32) -> two 16x16 f32 frags
      v8f sf0 = {}, sf1 = {};
      const int klo = hv ? 8 : 0;
#pragma unroll
      for (int step = 0; step < 2; ++step) {
        AF bk;
        int doff = step * 32 + klo;
        bk.q[0] = *(const uint4*)&kb[lh * 64 + doff];
        bk.q[1] = *(const uint4*)&kb[lh * 64 + doff + 16];
        sf0 = wmma_bf16(aq[step].v, bk.v, sf0);
        bk.q[0] = *(const uint4*)&kb[(16 + lh) * 64 + doff];
        bk.q[1] = *(const uint4*)&kb[(16 + lh) * 64 + doff + 16];
        sf1 = wmma_bf16(aq[step].v, bk.v, sf1);
      }

      // online softmax; row of element i = mrow + i + hv*8.
      // Row max must be lane-uniform (cross-lane max); row sum stays per-lane
      // partial (corr is lane-uniform) and is reduced once in the epilogue.
      bf16_t* pdst = &sP[wave * 512];
#pragma unroll
      for (int i = 0; i < 8; ++i) {
        int srow = mrow + i + hv * 8;
        float x0 = sf0[i] * scale; if (t0 + lh      > srow) x0 = -3.0e38f;
        float x1 = sf1[i] * scale; if (t0 + 16 + lh > srow) x1 = -3.0e38f;
        float cand = fmaxf(x0, x1);
        cand = fmaxf(cand, __shfl_xor(cand, 1, 32));
        cand = fmaxf(cand, __shfl_xor(cand, 2, 32));
        cand = fmaxf(cand, __shfl_xor(cand, 4, 32));
        cand = fmaxf(cand, __shfl_xor(cand, 8, 32));
        float mnew = fmaxf(mi[i], cand);
        float p0 = __expf(x0 - mnew);
        float p1 = __expf(x1 - mnew);
        float corr = __expf(mi[i] - mnew);
        li[i] = li[i] * corr + (p0 + p1);
        mi[i] = mnew;
        acc[0][i] *= corr; acc[1][i] *= corr;
        acc[2][i] *= corr; acc[3][i] *= corr;
        pdst[(i + hv * 8) * 32 + lh]      = (bf16_t)p0;
        pdst[(i + hv * 8) * 32 + 16 + lh] = (bf16_t)p1;
      }

      // reload P as A-fragment, then P(16x32) * V(32x64)
      AF pa;
      {
        int koff = hv ? 8 : 0;
        pa.q[0] = *(const uint4*)&pdst[lh * 32 + koff];
        pa.q[1] = *(const uint4*)&pdst[lh * 32 + koff + 16];
      }
#pragma unroll
      for (int j = 0; j < 4; ++j) {
        AF bv;
        bv.q[0] = *(const uint4*)&vb[(j * 16 + lh) * 32 + klo];
        bv.q[1] = *(const uint4*)&vb[(j * 16 + lh) * 32 + klo + 16];
        acc[j] = wmma_bf16(pa.v, bv.v, acc[j]);
      }
    }
    cur ^= 1;
  }

  // epilogue: reduce per-lane row-sum partials once, then write attn bf16
#pragma unroll
  for (int i = 0; i < 8; ++i) {
    float l = li[i];
    l += __shfl_xor(l, 1, 32);
    l += __shfl_xor(l, 2, 32);
    l += __shfl_xor(l, 4, 32);
    l += __shfl_xor(l, 8, 32);
    float inv = 1.0f / l;
    int srow = mrow + i + hv * 8;
    size_t base = ((size_t)b * SEQL + srow) * ATTD + (g * NH + head) * 64;
#pragma unroll
    for (int j = 0; j < 4; ++j)
      Out[base + j * 16 + lh] = (bf16_t)(acc[j][i] * inv);
  }
}

// ---------------- launch ----------------
extern "C" void kernel_launch(void* const* d_in, const int* in_sizes, int n_in,
                              void* d_out, int out_size, void* d_ws, size_t ws_size,
                              hipStream_t stream) {
  (void)in_sizes; (void)n_in; (void)out_size; (void)ws_size;
  const float* hidden  = (const float*)d_in[0];
  const float* cosb    = (const float*)d_in[1];
  const float* sinb    = (const float*)d_in[2];
  // d_in[3]=cu_seqlens, d_in[4]=max_s: shapes are compile-time constants here
  const float* qkv_w   = (const float*)d_in[5];
  const float* qkv_b   = (const float*)d_in[6];
  const float* dense_w = (const float*)d_in[7];
  const float* dense_b = (const float*)d_in[8];
  float* out = (float*)d_out;

  char* ws = (char*)d_ws;
  bf16_t* hid_bf    = (bf16_t*)(ws + 0);           // 16,777,216 B   [M][K]
  bf16_t* wqkv_bf   = (bf16_t*)(ws + 16777216);    // 12,582,912 B   [N][K] transposed
  bf16_t* wdense_bf = (bf16_t*)(ws + 29360128);    //  8,388,608 B   [N][K] transposed
  bf16_t* qkv_bf    = (bf16_t*)(ws + 37748736);    // 25,165,824 B
  bf16_t* Qb        = (bf16_t*)(ws + 62914560);    // 16,777,216 B
  bf16_t* Kb        = (bf16_t*)(ws + 79691776);    //  4,194,304 B
  bf16_t* Vtb       = (bf16_t*)(ws + 83886080);    //  4,194,304 B
  bf16_t* attn_bf   = (bf16_t*)(ws + 88080384);    // 16,777,216 B   [M][K]

  // 1) activations -> bf16; weights -> bf16 transposed [N][K]
  cvt_f32_bf16<<<(T_TOK * HID / 4 + 255) / 256, 256, 0, stream>>>(hidden, hid_bf,
                                                                  T_TOK * HID / 4);
  cvt_transpose<<<dim3(QKVN / 32, HID / 32), 256, 0, stream>>>(qkv_w, wqkv_bf,
                                                               HID, QKVN);
  cvt_transpose<<<dim3(ATTD / 32, HID / 32), 256, 0, stream>>>(dense_w, wdense_bf,
                                                               HID, ATTD);
  // 2) QKV projection (bf16 out)
  gemm_bias<1><<<dim3(QKVN / 128, T_TOK / 128), 256, 0, stream>>>(
      hid_bf, wqkv_bf, qkv_b, (void*)qkv_bf, T_TOK, QKVN, HID);
  // 3) RoPE + scatter
  rope_scatter<<<T_TOK, 256, 0, stream>>>(qkv_bf, cosb, sinb, Qb, Kb, Vtb);
  // 4) flash attention
  attn_kernel<<<4 * NG * (SEQL / 32), 256, 0, stream>>>(Qb, Kb, Vtb, attn_bf);
  // 5) output projection (f32 out, fused bias)
  gemm_bias<0><<<dim3(ATTD / 128, T_TOK / 128), 256, 0, stream>>>(
      attn_bf, wdense_bf, dense_b, (void*)out, T_TOK, ATTD, HID);
}
